// _MSA1_28664611734049
// MI455X (gfx1250) — matrix operand order repaired and checked by reading, not compiled, not run on hardware
//
#include <hip/hip_runtime.h>

// ---------------------------------------------------------------------------
// MSA window-3 attention for MI455X (gfx1250, wave32, WMMA).
//   K1: QKV projections  -> bf16 workspace   (WMMA bf16, f32 accum)
//   K2: window-3 attention core with the reference's reinterpreting reshapes
//   K3: output projection -> f32 d_out       (WMMA bf16, f32 accum)
// GEMM block tile: 256(M) x 128(N) x 32(K), 8 waves in a 4(M)x2(N) grid,
// each wave owns a 4x4 grid of 16x16 f32 accumulators (16 WMMA / k-step).
// LDS tiles are double-buffered: global loads for tile t+1 are issued before
// the 16 WMMAs of tile t, then converted/stored into the opposite buffer.
// ---------------------------------------------------------------------------

#define NHID_   512
#define NHEAD_  16
#define HDIM_   64
#define DMODEL_ 1024
#define BATCH_  8
#define LSEQ_   2048

#define ATILE_  (256 * 32)
#define BTILE_  (128 * 32)

typedef __attribute__((ext_vector_type(16))) __bf16 bf16x16;
typedef __attribute__((ext_vector_type(8)))  __bf16 bf16x8;
typedef __attribute__((ext_vector_type(8)))  float  f32x8;

// ---------------------------------------------------------------------------
// Tiled GEMM:  Out[n][m] = sum_k A[k][m] * B[k][n] + bias[n]
//   A: [K][M] (m contiguous), f32 or bf16.  B: [K][N] f32.  Out: [N][M].
// ---------------------------------------------------------------------------
template <int M, int N, int K, bool A_BF16, bool OUT_BF16>
__device__ __forceinline__ void gemm_block(
    const void* __restrict__ Aq, const float* __restrict__ Bw,
    const float* __restrict__ bias, void* __restrict__ Out,
    __bf16* As, __bf16* Bs) {
  const int tid   = threadIdx.x;
  const int lane  = tid & 31;
  const int wid   = tid >> 5;
  const int wm    = wid & 3;     // wave row: 64 m each
  const int wn    = wid >> 2;    // wave col: 64 n each
  const int khalf = lane >> 4;
  const int l15   = lane & 15;
  const int m0    = blockIdx.x * 256;
  const int n0    = blockIdx.y * 128;
  const int kiters = K / 32;

  f32x8 acc[4][4];
#pragma unroll
  for (int mi = 0; mi < 4; ++mi)
#pragma unroll
    for (int ni = 0; ni < 4; ++ni)
#pragma unroll
      for (int e = 0; e < 8; ++e) acc[mi][ni][e] = 0.0f;

  // stage registers for one tile of A and B
  float4 raf[8];   // used when A is f32
  bf16x8 rah[4];   // used when A is bf16
  float4 rb[4];

  // ---- issue + store tile 0
  {
    const int k0 = 0;
    if (A_BF16) {
      const __bf16* A = (const __bf16*)Aq;
#pragma unroll
      for (int i = 0; i < 4; ++i) {
        int idx = tid + i * 256, k = idx >> 5, m8 = (idx & 31) * 8;
        rah[i] = *(const bf16x8*)(A + (size_t)(k0 + k) * M + m0 + m8);
      }
    } else {
      const float* A = (const float*)Aq;
#pragma unroll
      for (int i = 0; i < 8; ++i) {
        int idx = tid + i * 256, k = idx >> 6, m4 = (idx & 63) * 4;
        raf[i] = *(const float4*)(A + (size_t)(k0 + k) * M + m0 + m4);
      }
    }
#pragma unroll
    for (int i = 0; i < 4; ++i) {
      int idx = tid + i * 256, k = idx >> 5, n4 = (idx & 31) * 4;
      rb[i] = *(const float4*)(Bw + (size_t)(k0 + k) * N + n0 + n4);
    }
    if (A_BF16) {
#pragma unroll
      for (int i = 0; i < 4; ++i) {
        int idx = tid + i * 256, k = idx >> 5, m8 = (idx & 31) * 8;
#pragma unroll
        for (int j = 0; j < 8; ++j) As[(m8 + j) * 32 + k] = rah[i][j];
      }
    } else {
#pragma unroll
      for (int i = 0; i < 8; ++i) {
        int idx = tid + i * 256, k = idx >> 6, m4 = (idx & 63) * 4;
        As[(m4 + 0) * 32 + k] = (__bf16)raf[i].x;
        As[(m4 + 1) * 32 + k] = (__bf16)raf[i].y;
        As[(m4 + 2) * 32 + k] = (__bf16)raf[i].z;
        As[(m4 + 3) * 32 + k] = (__bf16)raf[i].w;
      }
    }
#pragma unroll
    for (int i = 0; i < 4; ++i) {
      int idx = tid + i * 256, k = idx >> 5, n4 = (idx & 31) * 4;
      Bs[(n4 + 0) * 32 + k] = (__bf16)rb[i].x;
      Bs[(n4 + 1) * 32 + k] = (__bf16)rb[i].y;
      Bs[(n4 + 2) * 32 + k] = (__bf16)rb[i].z;
      Bs[(n4 + 3) * 32 + k] = (__bf16)rb[i].w;
    }
  }
  __syncthreads();

  int cur = 0;
  for (int t = 0; t < kiters; ++t) {
    // ---- issue global loads for tile t+1 (no wait yet)
    if (t + 1 < kiters) {
      const int k0 = (t + 1) * 32;
      if (A_BF16) {
        const __bf16* A = (const __bf16*)Aq;
#pragma unroll
        for (int i = 0; i < 4; ++i) {
          int idx = tid + i * 256, k = idx >> 5, m8 = (idx & 31) * 8;
          rah[i] = *(const bf16x8*)(A + (size_t)(k0 + k) * M + m0 + m8);
        }
      } else {
        const float* A = (const float*)Aq;
#pragma unroll
        for (int i = 0; i < 8; ++i) {
          int idx = tid + i * 256, k = idx >> 6, m4 = (idx & 63) * 4;
          raf[i] = *(const float4*)(A + (size_t)(k0 + k) * M + m0 + m4);
        }
      }
#pragma unroll
      for (int i = 0; i < 4; ++i) {
        int idx = tid + i * 256, k = idx >> 5, n4 = (idx & 31) * 4;
        rb[i] = *(const float4*)(Bw + (size_t)(k0 + k) * N + n0 + n4);
      }
    }
    // ---- prefetch tile t+2 into L2 (global_prefetch_b8)
    if (t + 2 < kiters) {
      size_t krow = (size_t)((t + 2) * 32 + (tid & 31));
      const char* pa = (const char*)Aq +
          (krow * M + m0 + (tid >> 5) * 32) * (A_BF16 ? 2 : 4);
      __builtin_prefetch(pa, 0, 1);
    }

    // ---- 16 WMMA for tile t out of buffer `cur`
    const __bf16* Ac = As + cur * ATILE_;
    const __bf16* Bc = Bs + cur * BTILE_;
    bf16x16 afrag[4];
#pragma unroll
    for (int mi = 0; mi < 4; ++mi) {
      const __bf16* p = &Ac[(wm * 64 + mi * 16 + l15) * 32 + 16 * khalf];
      bf16x8 lo = *(const bf16x8*)(const void*)p;          // K = 16k..16k+7
      bf16x8 hi = *(const bf16x8*)(const void*)(p + 8);    // K = 16k+8..
      afrag[mi] = __builtin_shufflevector(lo, hi, 0, 1, 2, 3, 4, 5, 6, 7, 8,
                                          9, 10, 11, 12, 13, 14, 15);
    }
#pragma unroll
    for (int ni = 0; ni < 4; ++ni) {
      const __bf16* p = &Bc[(wn * 64 + ni * 16 + l15) * 32 + 16 * khalf];
      bf16x16 bfrag = *(const bf16x16*)(const void*)p;     // K = 16k..16k+15
#pragma unroll
      for (int mi = 0; mi < 4; ++mi) {
        acc[mi][ni] = __builtin_amdgcn_wmma_f32_16x16x32_bf16(
            false, afrag[mi], false, bfrag, (short)0, acc[mi][ni], false,
            false);
      }
    }

    // ---- convert + store tile t+1 into the opposite buffer
    if (t + 1 < kiters) {
      __bf16* An = As + (cur ^ 1) * ATILE_;
      __bf16* Bn = Bs + (cur ^ 1) * BTILE_;
      if (A_BF16) {
#pragma unroll
        for (int i = 0; i < 4; ++i) {
          int idx = tid + i * 256, k = idx >> 5, m8 = (idx & 31) * 8;
#pragma unroll
          for (int j = 0; j < 8; ++j) An[(m8 + j) * 32 + k] = rah[i][j];
        }
      } else {
#pragma unroll
        for (int i = 0; i < 8; ++i) {
          int idx = tid + i * 256, k = idx >> 6, m4 = (idx & 63) * 4;
          An[(m4 + 0) * 32 + k] = (__bf16)raf[i].x;
          An[(m4 + 1) * 32 + k] = (__bf16)raf[i].y;
          An[(m4 + 2) * 32 + k] = (__bf16)raf[i].z;
          An[(m4 + 3) * 32 + k] = (__bf16)raf[i].w;
        }
      }
#pragma unroll
      for (int i = 0; i < 4; ++i) {
        int idx = tid + i * 256, k = idx >> 5, n4 = (idx & 31) * 4;
        Bn[(n4 + 0) * 32 + k] = (__bf16)rb[i].x;
        Bn[(n4 + 1) * 32 + k] = (__bf16)rb[i].y;
        Bn[(n4 + 2) * 32 + k] = (__bf16)rb[i].z;
        Bn[(n4 + 3) * 32 + k] = (__bf16)rb[i].w;
      }
      __syncthreads();
      cur ^= 1;
    }
  }

  // ---- epilogue: bias + store (contiguous 8 m-elements per lane)
#pragma unroll
  for (int ni = 0; ni < 4; ++ni) {
    int n = n0 + wn * 64 + ni * 16 + l15;
    float bvl = bias[n];
#pragma unroll
    for (int mi = 0; mi < 4; ++mi) {
      int m = m0 + wm * 64 + mi * 16 + 8 * khalf;
      f32x8 a = acc[mi][ni];
      if (OUT_BF16) {
        __bf16* O = (__bf16*)Out;
        bf16x8 o;
#pragma unroll
        for (int e = 0; e < 8; ++e) o[e] = (__bf16)(a[e] + bvl);
        *(bf16x8*)(O + (size_t)n * M + m) = o;
      } else {
        float* O = (float*)Out;
        *(float4*)(O + (size_t)n * M + m) =
            make_float4(a[0] + bvl, a[1] + bvl, a[2] + bvl, a[3] + bvl);
        *(float4*)(O + (size_t)n * M + m + 4) =
            make_float4(a[4] + bvl, a[5] + bvl, a[6] + bvl, a[7] + bvl);
      }
    }
  }
}

// K1: q/k/v = x^T W + b  (one of 3 selected by blockIdx.z%3), bf16 out.
__global__ __launch_bounds__(256) void qkv_gemm_kernel(
    const float* __restrict__ x, const float* __restrict__ wq,
    const float* __restrict__ bq, const float* __restrict__ wk,
    const float* __restrict__ bk, const float* __restrict__ wv,
    const float* __restrict__ bv, __bf16* __restrict__ qkv) {
  __shared__ __align__(32) __bf16 As[2 * ATILE_];
  __shared__ __align__(32) __bf16 Bs[2 * BTILE_];
  const int z = blockIdx.z;
  const int b = z / 3, s = z - 3 * b;
  const float* W  = (s == 0) ? wq : ((s == 1) ? wk : wv);
  const float* Bi = (s == 0) ? bq : ((s == 1) ? bk : bv);
  const float* A  = x + (size_t)b * NHID_ * LSEQ_;
  __bf16* Out = qkv + ((size_t)s * BATCH_ + b) * (size_t)DMODEL_ * LSEQ_;
  gemm_block<LSEQ_, DMODEL_, NHID_, false, true>(A, W, Bi, Out, As, Bs);
}

// K3: out = att2^T wo + bo, f32 out.
__global__ __launch_bounds__(256) void out_gemm_kernel(
    const __bf16* __restrict__ att2, const float* __restrict__ wo,
    const float* __restrict__ bo, float* __restrict__ out) {
  __shared__ __align__(32) __bf16 As[2 * ATILE_];
  __shared__ __align__(32) __bf16 Bs[2 * BTILE_];
  const int b = blockIdx.z;
  gemm_block<LSEQ_, NHID_, DMODEL_, true, false>(
      att2 + (size_t)b * DMODEL_ * LSEQ_, wo, bo,
      out + (size_t)b * NHID_ * LSEQ_, As, Bs);
}

// ---------------------------------------------------------------------------
// K2: window-3 attention with the reference's reinterpreting reshapes.
//   unfold scramble:  (c',w') -> w=(3c'+w')>>10, c=(3c'+w')&1023, l_src=l+w-1
//   output scramble:  att[b,h,hd,l] -> att2[b, h*64+(l>>5), (l&31)*64+hd]
//   (each thread's 64 hd outputs are 64 contiguous bf16 -> coalesced stores)
// ---------------------------------------------------------------------------
__global__ __launch_bounds__(256) void attn_kernel(
    const __bf16* __restrict__ q, const __bf16* __restrict__ k,
    const __bf16* __restrict__ v, __bf16* __restrict__ att2) {
  const int g = blockIdx.x * 256 + threadIdx.x;  // B*NHEAD*L threads
  const int l = g & (LSEQ_ - 1);
  const int h = (g >> 11) & (NHEAD_ - 1);
  const int b = g >> 15;
  const size_t base = (size_t)b * DMODEL_ * LSEQ_;

  // pass 1: window scores  s[w'] = sum_hd q * k_unf / sqrt(hd)
  float s0 = 0.f, s1 = 0.f, s2 = 0.f;
#pragma unroll 4
  for (int hd = 0; hd < HDIM_; ++hd) {
    const int cq = h * HDIM_ + hd;
    const float qv = (float)q[base + (size_t)cq * LSEQ_ + l];
#pragma unroll
    for (int w = 0; w < 3; ++w) {
      const int flat = cq * 3 + w;
      const int wsh  = flat >> 10;       // window shift 0..2
      const int c    = flat & 1023;      // source channel
      const int ls   = l + wsh - 1;
      const float kv = (ls >= 0 && ls < LSEQ_)
                           ? (float)k[base + (size_t)c * LSEQ_ + ls]
                           : 0.f;
      if (w == 0) s0 += qv * kv;
      else if (w == 1) s1 += qv * kv;
      else s2 += qv * kv;
    }
  }
  const float scale = 0.125f;  // 1/sqrt(64)
  s0 *= scale; s1 *= scale; s2 *= scale;
  const float mx = fmaxf(s0, fmaxf(s1, s2));
  const float e0 = __expf(s0 - mx), e1 = __expf(s1 - mx), e2 = __expf(s2 - mx);
  const float inv = 1.f / (e0 + e1 + e2);
  const float a0 = e0 * inv, a1 = e1 * inv, a2 = e2 * inv;

  // pass 2: att = sum_w alpha*v_unf, scattered per the output reshape
  const int dout = h * HDIM_ + (l >> 5);
  __bf16* op = att2 + base + (size_t)dout * LSEQ_ + (size_t)(l & 31) * HDIM_;
  bf16x8 ov;
#pragma unroll 8
  for (int hd = 0; hd < HDIM_; ++hd) {
    const int cq = h * HDIM_ + hd;
    float acc = 0.f;
#pragma unroll
    for (int w = 0; w < 3; ++w) {
      const int flat = cq * 3 + w;
      const int wsh  = flat >> 10;
      const int c    = flat & 1023;
      const int ls   = l + wsh - 1;
      const float vv = (ls >= 0 && ls < LSEQ_)
                           ? (float)v[base + (size_t)c * LSEQ_ + ls]
                           : 0.f;
      acc += (w == 0 ? a0 : (w == 1 ? a1 : a2)) * vv;
    }
    ov[hd & 7] = (__bf16)acc;
    if ((hd & 7) == 7) *(bf16x8*)(op + (hd & ~7)) = ov;
  }
}

// ---------------------------------------------------------------------------
extern "C" void kernel_launch(void* const* d_in, const int* in_sizes, int n_in,
                              void* d_out, int out_size, void* d_ws,
                              size_t ws_size, hipStream_t stream) {
  (void)in_sizes; (void)n_in; (void)out_size; (void)ws_size;
  const float* x  = (const float*)d_in[0];
  const float* wq = (const float*)d_in[1];
  const float* bq = (const float*)d_in[2];
  const float* wk = (const float*)d_in[3];
  const float* bk = (const float*)d_in[4];
  const float* wv = (const float*)d_in[5];
  const float* bv = (const float*)d_in[6];
  const float* wo = (const float*)d_in[7];
  const float* bo = (const float*)d_in[8];
  float* out = (float*)d_out;

  // ws (bf16): Q | K | V | ATT2, each B*D*L elems  (total 134 MB)
  const size_t BDL = (size_t)BATCH_ * DMODEL_ * LSEQ_;
  __bf16* ws   = (__bf16*)d_ws;
  __bf16* qb   = ws;
  __bf16* kb   = ws + BDL;
  __bf16* vb   = ws + 2 * BDL;
  __bf16* att2 = ws + 3 * BDL;

  qkv_gemm_kernel<<<dim3(LSEQ_ / 256, DMODEL_ / 128, BATCH_ * 3), 256, 0,
                    stream>>>(x, wq, bq, wk, bk, wv, bv, ws);
  attn_kernel<<<dim3((BATCH_ * NHEAD_ * LSEQ_) / 256), 256, 0, stream>>>(
      qb, kb, vb, att2);
  out_gemm_kernel<<<dim3(LSEQ_ / 256, NHID_ / 128, BATCH_), 256, 0, stream>>>(
      att2, wo, bo, out);
}
